// BDFR_80925773791448
// MI455X (gfx1250) — matrix operand order repaired and checked by reading, not compiled, not run on hardware
//
#include <hip/hip_runtime.h>
#include <hip/hip_bf16.h>
#include <math.h>

typedef __attribute__((ext_vector_type(16))) _Float16 v16h;
typedef __attribute__((ext_vector_type(8)))  float    v8f;

#define HH 128
#define WW 128
#define HWP (HH*WW)          // 16384
#define CIN 256
#define PCH 64
#define KPR 8
#define NIMG 4

__device__ __forceinline__ float silu_f(float v) {
    return v / (1.0f + __expf(-v));
}

// ---------------------------------------------------------------------------
// Kernel 0: init per-image min/max slots (int-bit atomics; min_dist >= 0)
// ---------------------------------------------------------------------------
__global__ void init_minmax(int* __restrict__ mm) {
    int t = threadIdx.x;
    if (t < NIMG)                mm[t] = 0x7F800000;  // +inf
    else if (t < 2 * NIMG)       mm[t] = 0;           // max over nonneg values
}

// ---------------------------------------------------------------------------
// Kernel 1: 1x1 conv (GEMM M=64,K=256,N=B*HW) + BN + SiLU via WMMA f16->f32.
// Block = 256 px of one image, 8 waves = 4 M-tiles x 2 pixel halves (128 px).
// ---------------------------------------------------------------------------
__global__ __launch_bounds__(256) void conv1x1_wmma(
    const float* __restrict__ x,      // (B, 256, HW)
    const float* __restrict__ w1,     // (64, 256)
    const float* __restrict__ bn1,    // (4, 64)
    float* __restrict__ h_out)        // (B, 64, HW)
{
    const int wave  = threadIdx.x >> 5;
    const int lane  = threadIdx.x & 31;
    const int mtile = wave & 3;           // 4 M-tiles of 16 out-channels
    const int nhalf = wave >> 2;          // which 128-px half of the block

    const int pix0  = blockIdx.x * 256;   // blocks never straddle images
    const int b     = pix0 / HWP;
    const int pbase = (pix0 - b * HWP) + nhalf * 128;

    const int lo = lane & 15;
    const int hi = lane >> 4;

    // --- A fragments: weight rows m = mtile*16 + lo, 8 K-steps of 32 ---
    // 16-bit A layout (ISA 7.12.2): k = kstep*32 + (j<8?0:16) + hi*8 + (j&7)
    const float* wrow = w1 + (size_t)(mtile * 16 + lo) * CIN;
    v16h afrag[8];
#pragma unroll
    for (int kk = 0; kk < 8; ++kk) {
#pragma unroll
        for (int j = 0; j < 16; ++j) {
            int k = kk * 32 + ((j < 8) ? 0 : 16) + hi * 8 + (j & 7);
            afrag[kk][j] = (_Float16)wrow[k];
        }
    }

    // --- BN params for this lane's 8 accumulator rows: m = mtile*16 + r + 8*hi
    float bsc[8], bsh[8];
#pragma unroll
    for (int r = 0; r < 8; ++r) {
        int m = mtile * 16 + r + 8 * hi;
        float g  = bn1[0 * PCH + m];
        float be = bn1[1 * PCH + m];
        float mu = bn1[2 * PCH + m];
        float va = bn1[3 * PCH + m];
        float s  = g * rsqrtf(va + 1e-5f);
        bsc[r] = s;
        bsh[r] = be - mu * s;
    }

    const float* xb = x + (size_t)b * CIN * HWP;

    for (int nt = 0; nt < 8; ++nt) {
        const int n0 = pbase + nt * 16;
        v8f acc = {};
#pragma unroll
        for (int kk = 0; kk < 8; ++kk) {
            // 16-bit B layout (K x N): same k order as A: k = (j<8?0:16) + hi*8 + (j&7)
            v16h bfrag;
#pragma unroll
            for (int j = 0; j < 16; ++j) {
                int c = kk * 32 + ((j < 8) ? 0 : 16) + hi * 8 + (j & 7);
                bfrag[j] = (_Float16)xb[(size_t)c * HWP + n0 + lo];
            }
            acc = __builtin_amdgcn_wmma_f32_16x16x32_f16(
                false, afrag[kk], false, bfrag, (short)0, acc, false, false);
        }
        // epilogue: BN + SiLU, D layout: row m = r + 8*hi, col n = lane%16
#pragma unroll
        for (int r = 0; r < 8; ++r) {
            int m = mtile * 16 + r + 8 * hi;
            float v = acc[r] * bsc[r] + bsh[r];
            h_out[((size_t)b * PCH + m) * HWP + n0 + lo] = silu_f(v);
        }
    }
}

// ---------------------------------------------------------------------------
// Kernel 2: depthwise 3x3 + BN + SiLU -> proj; per-pixel min L2 distance to 8
// prototypes; per-image min/max via int atomics. Block = 16x16 tile, 1 image.
// ---------------------------------------------------------------------------
__global__ __launch_bounds__(256) void dw_dist(
    const float* __restrict__ h,       // (B, 64, HW)
    const float* __restrict__ dwW,     // (64, 1, 3, 3)
    const float* __restrict__ bn2,     // (4, 64)
    const float* __restrict__ protos,  // (8, 64)
    float* __restrict__ mind,          // (B, HW)
    int* __restrict__ mm)              // [min[4], max[4]]
{
    __shared__ float tile[18 * 20];
    __shared__ float wts[64 * 9];
    __shared__ float sc[64], sh[64];
    __shared__ float pr[8 * 64];
    __shared__ float psq[8];
    __shared__ float redmin[8], redmax[8];

    const int t   = threadIdx.x;
    const int b   = blockIdx.x >> 6;
    const int tl  = blockIdx.x & 63;
    const int ty0 = (tl >> 3) * 16;
    const int tx0 = (tl & 7) * 16;

    for (int i = t; i < 576; i += 256) wts[i] = dwW[i];
    for (int i = t; i < 512; i += 256) pr[i]  = protos[i];
    if (t < 64) {
        float g  = bn2[t], be = bn2[64 + t], mu = bn2[128 + t], va = bn2[192 + t];
        float s  = g * rsqrtf(va + 1e-5f);
        sc[t] = s; sh[t] = be - mu * s;
    }
    __syncthreads();
    if (t < 8) {
        float s = 0.f;
        for (int c = 0; c < 64; ++c) { float p = pr[t * 64 + c]; s += p * p; }
        psq[t] = s;
    }
    __syncthreads();

    const int py = t >> 4, px = t & 15;
    const int gy = ty0 + py, gx = tx0 + px;

    float sumsq = 0.f;
    float dots[8] = {0.f, 0.f, 0.f, 0.f, 0.f, 0.f, 0.f, 0.f};

    for (int c = 0; c < 64; ++c) {
        const float* hc = h + ((size_t)b * PCH + c) * HWP;
        for (int i = t; i < 324; i += 256) {
            int ly = i / 18, lx = i - ly * 18;
            int sy = ty0 + ly - 1, sx = tx0 + lx - 1;
            float v = 0.f;
            if (sy >= 0 && sy < HH && sx >= 0 && sx < WW) v = hc[sy * WW + sx];
            tile[ly * 20 + lx] = v;
        }
        __syncthreads();
        float acc = 0.f;
        const float* w = &wts[c * 9];
#pragma unroll
        for (int dy = 0; dy < 3; ++dy)
#pragma unroll
            for (int dx = 0; dx < 3; ++dx)
                acc += w[dy * 3 + dx] * tile[(py + dy) * 20 + (px + dx)];
        float p = silu_f(acc * sc[c] + sh[c]);
        sumsq += p * p;
#pragma unroll
        for (int k = 0; k < 8; ++k) dots[k] += p * pr[k * 64 + c];
        __syncthreads();
    }

    float md = 1e30f;
#pragma unroll
    for (int k = 0; k < 8; ++k) {
        float d2 = sumsq + psq[k] - 2.f * dots[k];
        md = fminf(md, sqrtf(fmaxf(d2, 0.f)));
    }
    md = md / (1.0f + 1e-6f);   // / (TAU + 1e-6)
    mind[(size_t)b * HWP + gy * WW + gx] = md;

    // block min/max reduction (wave32 shuffles + LDS across 8 waves)
    float mn = md, mx = md;
#pragma unroll
    for (int o = 16; o > 0; o >>= 1) {
        mn = fminf(mn, __shfl_xor(mn, o));
        mx = fmaxf(mx, __shfl_xor(mx, o));
    }
    if ((t & 31) == 0) { redmin[t >> 5] = mn; redmax[t >> 5] = mx; }
    __syncthreads();
    if (t == 0) {
        for (int i = 1; i < 8; ++i) {
            mn = fminf(mn, redmin[i]);
            mx = fmaxf(mx, redmax[i]);
        }
        atomicMin(&mm[b],        __float_as_int(mn));
        atomicMax(&mm[NIMG + b], __float_as_int(mx));
    }
}

// ---------------------------------------------------------------------------
// Kernel 3: deviation -> two DS-conv branches -> fuse -> sigmoid -> attn map
// ---------------------------------------------------------------------------
__global__ __launch_bounds__(256) void attn_k(
    const float* __restrict__ mind, const int* __restrict__ mm,
    const float* __restrict__ bs_dw, const float* __restrict__ bs_bn1,
    const float* __restrict__ bs_pw, const float* __restrict__ bs_bn2,
    const float* __restrict__ bl_dw, const float* __restrict__ bl_bn1,
    const float* __restrict__ bl_pw, const float* __restrict__ bl_bn2,
    const float* __restrict__ fuse_w, const float* __restrict__ fuse_b,
    float* __restrict__ attn)
{
    __shared__ float dev[20 * 22];
    const int t   = threadIdx.x;
    const int b   = blockIdx.x >> 6;
    const int tl  = blockIdx.x & 63;
    const int ty0 = (tl >> 3) * 16;
    const int tx0 = (tl & 7) * 16;

    const float dmin = __int_as_float(mm[b]);
    const float dmax = __int_as_float(mm[NIMG + b]);
    const float inv  = 1.0f / (dmax - dmin + 1e-6f);
    const float* mb  = mind + (size_t)b * HWP;

    for (int i = t; i < 400; i += 256) {
        int ly = i / 20, lx = i - ly * 20;
        int sy = ty0 + ly - 2, sx = tx0 + lx - 2;
        float v = 0.f;   // zero-padded deviation outside image
        if (sy >= 0 && sy < HH && sx >= 0 && sx < WW)
            v = (mb[sy * WW + sx] - dmin) * inv;
        dev[ly * 22 + lx] = v;
    }
    __syncthreads();

    const int py = t >> 4, px = t & 15;

    // branch 1: depthwise 3x3 (center at py+2, px+2 in 20x22 halo frame)
    float a1 = 0.f;
#pragma unroll
    for (int dy = 0; dy < 3; ++dy)
#pragma unroll
        for (int dx = 0; dx < 3; ++dx)
            a1 += bs_dw[dy * 3 + dx] * dev[(py + 1 + dy) * 22 + (px + 1 + dx)];
    {
        float s1 = bs_bn1[0] * rsqrtf(bs_bn1[3] + 1e-5f);
        a1 = silu_f(a1 * s1 + (bs_bn1[1] - bs_bn1[2] * s1));
    }

    // branch 2: depthwise 5x5
    float a2 = 0.f;
#pragma unroll
    for (int dy = 0; dy < 5; ++dy)
#pragma unroll
        for (int dx = 0; dx < 5; ++dx)
            a2 += bl_dw[dy * 5 + dx] * dev[(py + dy) * 22 + (px + dx)];
    {
        float s2 = bl_bn1[0] * rsqrtf(bl_bn1[3] + 1e-5f);
        a2 = silu_f(a2 * s2 + (bl_bn1[1] - bl_bn1[2] * s2));
    }

    // pointwise 1->8 per branch, BN + SiLU, fused 1x1 (16->1)
    float acc = fuse_b[0];
#pragma unroll
    for (int k = 0; k < 8; ++k) {
        float s = bs_bn2[k] * rsqrtf(bs_bn2[24 + k] + 1e-5f);
        float v = silu_f(bs_pw[k] * a1 * s + (bs_bn2[8 + k] - bs_bn2[16 + k] * s));
        acc += fuse_w[k] * v;
        float s5 = bl_bn2[k] * rsqrtf(bl_bn2[24 + k] + 1e-5f);
        float v5 = silu_f(bl_pw[k] * a2 * s5 + (bl_bn2[8 + k] - bl_bn2[16 + k] * s5));
        acc += fuse_w[8 + k] * v5;
    }
    float a = 1.0f / (1.0f + __expf(-acc));
    attn[(size_t)b * HWP + (ty0 + py) * WW + (tx0 + px)] = a;
}

// ---------------------------------------------------------------------------
// Kernel 4: out = x * (1 + gamma * attn)   (float4 vectorized, HBM-bound)
// ---------------------------------------------------------------------------
__global__ __launch_bounds__(256) void apply_k(
    const float* __restrict__ x, const float* __restrict__ attn,
    const float* __restrict__ gamma, float* __restrict__ out)
{
    const size_t i = ((size_t)blockIdx.x * blockDim.x + threadIdx.x) * 4;
    const float g  = gamma[0];
    const size_t pix = i & (size_t)(HWP - 1);
    const size_t b   = (i >> 14) >> 8;   // i / (HW * 256)

    const float4 xv = *(const float4*)(x + i);
    const float4 av = *(const float4*)(attn + b * HWP + pix);
    float4 o;
    o.x = xv.x * (1.0f + g * av.x);
    o.y = xv.y * (1.0f + g * av.y);
    o.z = xv.z * (1.0f + g * av.z);
    o.w = xv.w * (1.0f + g * av.w);
    *(float4*)(out + i) = o;
}

// ---------------------------------------------------------------------------
extern "C" void kernel_launch(void* const* d_in, const int* in_sizes, int n_in,
                              void* d_out, int out_size, void* d_ws, size_t ws_size,
                              hipStream_t stream) {
    const float* x      = (const float*)d_in[0];
    const float* fp_w1  = (const float*)d_in[1];
    const float* fp_bn1 = (const float*)d_in[2];
    const float* fp_dw  = (const float*)d_in[3];
    const float* fp_bn2 = (const float*)d_in[4];
    const float* protos = (const float*)d_in[5];
    const float* bs_dw  = (const float*)d_in[6];
    const float* bs_bn1 = (const float*)d_in[7];
    const float* bs_pw  = (const float*)d_in[8];
    const float* bs_bn2 = (const float*)d_in[9];
    const float* bl_dw  = (const float*)d_in[10];
    const float* bl_bn1 = (const float*)d_in[11];
    const float* bl_pw  = (const float*)d_in[12];
    const float* bl_bn2 = (const float*)d_in[13];
    const float* fuse_w = (const float*)d_in[14];
    const float* fuse_b = (const float*)d_in[15];
    const float* gamma  = (const float*)d_in[16];

    // workspace layout (floats): [minmax int 8][min_dist B*HW][attn B*HW][h B*64*HW]
    float* ws   = (float*)d_ws;
    int*   mm   = (int*)d_ws;
    float* mind = ws + 8;
    float* attn = ws + 8 + NIMG * HWP;
    float* h    = ws + 8 + 2 * NIMG * HWP;

    init_minmax<<<1, 32, 0, stream>>>(mm);
    conv1x1_wmma<<<NIMG * HWP / 256, 256, 0, stream>>>(x, fp_w1, fp_bn1, h);
    dw_dist<<<NIMG * 64, 256, 0, stream>>>(h, fp_dw, fp_bn2, protos, mind, mm);
    attn_k<<<NIMG * 64, 256, 0, stream>>>(mind, mm,
                                          bs_dw, bs_bn1, bs_pw, bs_bn2,
                                          bl_dw, bl_bn1, bl_pw, bl_bn2,
                                          fuse_w, fuse_b, attn);
    apply_k<<<(NIMG * CIN * HWP) / (256 * 4), 256, 0, stream>>>(x, attn, gamma,
                                                                (float*)d_out);
}
